// PMSPGNN_60988535603417
// MI455X (gfx1250) — matrix operand order repaired and checked by reading, not compiled, not run on hardware
//
#include <hip/hip_runtime.h>
#include <math.h>

// ---------------------------------------------------------------------------
// CDNA5 / gfx1250 types for WMMA bf16 (wave32)
// ---------------------------------------------------------------------------
typedef __bf16 bf16_t;
typedef __attribute__((ext_vector_type(16))) __bf16 v16bf;
typedef __attribute__((ext_vector_type(8)))  __bf16 v8bf;
typedef __attribute__((ext_vector_type(8)))  float  v8f;

#define GF_APPLY  2          // apply bias + activation on store
#define GF_LEAKY  (1 << 2)
#define GF_RELU   (2 << 2)

// One K-segment of the (virtually concatenated) A operand.
struct Seg {
    const float* A;      // f32 activations
    const int*   ridx;   // optional row gather (nullptr = identity)
    int rstride;         // gather index stride (edge_index is [E][2])
    int lda;             // A row stride in floats
    int K;               // segment width
};

__device__ __forceinline__ v16bf frag_a(const bf16_t* row, int sub)
{
    // A fragment (16x32, ISA 7.12.2): lanes 0-15 hold K 0-7 & 16-23,
    // lanes 16-31 hold K 8-15 & 24-31.
    const v8bf lo = *(const v8bf*)(row + sub * 8);
    const v8bf hi = *(const v8bf*)(row + 16 + sub * 8);
    return __builtin_shufflevector(lo, hi,
        0,1,2,3,4,5,6,7,8,9,10,11,12,13,14,15);
}

__device__ __forceinline__ v16bf frag_b(const bf16_t* col, int sub)
{
    // B fragment (32x16): lanes 0-15 hold K 0-7 & 16-23, lanes 16-31 hold K 8-15 & 24-31.
    const v8bf lo = *(const v8bf*)(col + sub * 8);
    const v8bf hi = *(const v8bf*)(col + 16 + sub * 8);
    return __builtin_shufflevector(lo, hi,
        0,1,2,3,4,5,6,7,8,9,10,11,12,13,14,15);
}

// ---------------------------------------------------------------------------
// Fused multi-segment GEMM:
//   C[M x N] = act( [seg0 | seg1 | seg2] @ W[(K0+K1+K2) x N] + bias )
// - A rows gathered per segment (fuses x[src]/x[tgt] concat)
// - f32 -> bf16 conversion during LDS staging; full-K tiles take a branch-free
//   vectorized b128 path, the single K=16 GEMM takes the peeled tail tile
// - block tile 128(M) x 64(N), 8 waves (4x2); each wave computes a 2x2 grid of
//   16x16 tiles (2 A-frags x 2 B-frags) -> 4x v_wmma_f32_16x16x32_bf16 / step
// - M % 128 == 0, N % 64 == 0 (true for all launched shapes)
// ---------------------------------------------------------------------------
__global__ __launch_bounds__(256)
void gemm_bf16_wmma(Seg s0, Seg s1, Seg s2, int nseg,
                    const float* __restrict__ W,
                    const float* __restrict__ bias,
                    float* __restrict__ C,
                    int M, int N, int flags)
{
    __shared__ bf16_t As[128][32];  // [row][k]
    __shared__ bf16_t Bs[64][32];   // [n][k] (W tile transposed)

    const int tid  = threadIdx.x;
    const int wave = tid >> 5;
    const int lane = tid & 31;
    const int mw   = wave >> 1;       // 0..3  (two 16-row slices: mw*16, 64+mw*16)
    const int nw   = wave & 1;        // 0..1  (32-col slice -> two 16-col tiles)
    const int sub  = lane >> 4;       // lane half
    const int idx  = lane & 15;

    const int blockM = blockIdx.x * 128;
    const int blockN = blockIdx.y * 64;

    // staging thread mapping: 8 elems per thread per 64-row half
    const int str = tid >> 2;          // 0..63
    const int stc = (tid & 3) * 8;     // 0,8,16,24

    v8f acc00, acc01, acc10, acc11;
#pragma unroll
    for (int r = 0; r < 8; ++r) {
        acc00[r] = 0.0f; acc01[r] = 0.0f; acc10[r] = 0.0f; acc11[r] = 0.0f;
    }

    const int ar  = mw * 16 + idx;
    const int bn0 = nw * 32 + idx;

    auto mma_step = [&]() {
        __syncthreads();                       // staging visible to all waves
        const v16bf af0 = frag_a(&As[ar][0], sub);
        const v16bf af1 = frag_a(&As[64 + ar][0], sub);
        const v16bf bf0 = frag_b(&Bs[bn0][0], sub);
        const v16bf bf1 = frag_b(&Bs[bn0 + 16][0], sub);
        acc00 = __builtin_amdgcn_wmma_f32_16x16x32_bf16(
            false, af0, false, bf0, (short)0, acc00, false, false);
        acc01 = __builtin_amdgcn_wmma_f32_16x16x32_bf16(
            false, af0, false, bf1, (short)0, acc01, false, false);
        acc10 = __builtin_amdgcn_wmma_f32_16x16x32_bf16(
            false, af1, false, bf0, (short)0, acc10, false, false);
        acc11 = __builtin_amdgcn_wmma_f32_16x16x32_bf16(
            false, af1, false, bf1, (short)0, acc11, false, false);
    };

    const float* Wseg = W;
    for (int sidx = 0; sidx < nseg; ++sidx) {
        const Seg sg = (sidx == 0) ? s0 : (sidx == 1) ? s1 : s2;

        // hoist gather + A base pointers (both 64-row halves) out of K loop
        const int grow0 = blockM + str;
        const int grow1 = grow0 + 64;
        const int arow0 = sg.ridx ? sg.ridx[(size_t)grow0 * sg.rstride] : grow0;
        const int arow1 = sg.ridx ? sg.ridx[(size_t)grow1 * sg.rstride] : grow1;
        const float* __restrict__ ap0 = sg.A + (size_t)arow0 * sg.lda;
        const float* __restrict__ ap1 = sg.A + (size_t)arow1 * sg.lda;

        const int kfull = sg.K & ~31;
        int k0 = 0;
        // ---------------- branch-free full-tile loop ----------------
        for (; k0 < kfull; k0 += 32) {
            __syncthreads();                   // fragments consumed
            {   // A tile rows [0,64) and [64,128): b128 loads + pack
                const float4 f0 = *(const float4*)(ap0 + k0 + stc);
                const float4 f1 = *(const float4*)(ap0 + k0 + stc + 4);
                const float4 g0 = *(const float4*)(ap1 + k0 + stc);
                const float4 g1 = *(const float4*)(ap1 + k0 + stc + 4);
                v8bf p, q;
                p[0] = (bf16_t)f0.x; p[1] = (bf16_t)f0.y;
                p[2] = (bf16_t)f0.z; p[3] = (bf16_t)f0.w;
                p[4] = (bf16_t)f1.x; p[5] = (bf16_t)f1.y;
                p[6] = (bf16_t)f1.z; p[7] = (bf16_t)f1.w;
                q[0] = (bf16_t)g0.x; q[1] = (bf16_t)g0.y;
                q[2] = (bf16_t)g0.z; q[3] = (bf16_t)g0.w;
                q[4] = (bf16_t)g1.x; q[5] = (bf16_t)g1.y;
                q[6] = (bf16_t)g1.z; q[7] = (bf16_t)g1.w;
                *(v8bf*)&As[str][stc]      = p;   // ds_store_b128
                *(v8bf*)&As[64 + str][stc] = q;
            }
            {   // B tile transposed: Bs[n][k]
                const float* __restrict__ wp =
                    Wseg + (size_t)(k0 + stc) * N + blockN + str;
                v8bf p;
#pragma unroll
                for (int j = 0; j < 8; ++j)
                    p[j] = (bf16_t)wp[(size_t)j * N];
                *(v8bf*)&Bs[str][stc] = p;
            }
            mma_step();
        }
        // ---------------- peeled K tail (only K=16 GEMM) ----------------
        if (k0 < sg.K) {
            __syncthreads();
            {
                v8bf p, q;
#pragma unroll
                for (int j = 0; j < 8; ++j) {
                    const int kg = k0 + stc + j;
                    const bool in = kg < sg.K;
                    p[j] = (bf16_t)(in ? ap0[kg] : 0.0f);
                    q[j] = (bf16_t)(in ? ap1[kg] : 0.0f);
                }
                *(v8bf*)&As[str][stc]      = p;
                *(v8bf*)&As[64 + str][stc] = q;
            }
            {
                const float* __restrict__ wp =
                    Wseg + (size_t)(k0 + stc) * N + blockN + str;
                v8bf p;
#pragma unroll
                for (int j = 0; j < 8; ++j) {
                    const int kg = k0 + stc + j;
                    p[j] = (bf16_t)((kg < sg.K) ? wp[(size_t)j * N] : 0.0f);
                }
                *(v8bf*)&Bs[str][stc] = p;
            }
            mma_step();
        }
        Wseg += (size_t)sg.K * N;
    }

    // ---- epilogue: bias + activation for the 2x2 tile grid ----
    const int rowC0 = blockM + mw * 16 + sub * 8;
    const int rowC1 = rowC0 + 64;
    const int colC0 = blockN + nw * 32 + idx;
    const int colC1 = colC0 + 16;
    const int act   = flags >> 2;
    float b0 = 0.0f, b1 = 0.0f;
    if ((flags & GF_APPLY) && bias) { b0 = bias[colC0]; b1 = bias[colC1]; }

    auto activate = [&](float v) {
        if (flags & GF_APPLY) {
            if (act == 1)      v = (v > 0.0f) ? v : 0.2f * v;
            else if (act == 2) v = fmaxf(v, 0.0f);
        }
        return v;
    };
#pragma unroll
    for (int r = 0; r < 8; ++r) {
        C[(size_t)(rowC0 + r) * N + colC0] = activate(acc00[r] + b0);
        C[(size_t)(rowC0 + r) * N + colC1] = activate(acc01[r] + b1);
        C[(size_t)(rowC1 + r) * N + colC0] = activate(acc10[r] + b0);
        C[(size_t)(rowC1 + r) * N + colC1] = activate(acc11[r] + b1);
    }
}

// ---------------------------------------------------------------------------
// 1-row MLP layer / u-bias fold:  out[j] = act( sum_k h[k]*W[k*N+j] + b[j] )
// ---------------------------------------------------------------------------
__global__ void tiny_mlp(const float* __restrict__ h, int K,
                         const float* __restrict__ W,
                         const float* __restrict__ b,
                         float* __restrict__ out, int N, int act)
{
    const int j = blockIdx.x * blockDim.x + threadIdx.x;
    if (j >= N) return;
    float s = b ? b[j] : 0.0f;
    for (int k = 0; k < K; ++k) s += h[k] * W[(size_t)k * N + j];
    if (act == 1)      s = (s > 0.0f) ? s : 0.2f * s;
    else if (act == 2) s = (s > 0.0f) ? s : 0.0f;
    out[j] = s;
}

__global__ void zero_f32(float* __restrict__ p, size_t n)
{
    size_t i = (size_t)blockIdx.x * blockDim.x + threadIdx.x;
    size_t stride = (size_t)gridDim.x * blockDim.x;
    for (; i < n; i += stride) p[i] = 0.0f;
}

__global__ void copy_f32(const float* __restrict__ s, float* __restrict__ d, int n)
{
    const int i = blockIdx.x * blockDim.x + threadIdx.x;
    if (i < n) d[i] = s[i];
}

// column sum: out[j] = sum_i A[i*C+j]
__global__ void colsum(const float* __restrict__ A, int M, int C,
                       float* __restrict__ out)
{
    const int j = blockIdx.x * blockDim.x + threadIdx.x;
    if (j >= C) return;
    float s = 0.0f;
    for (int i = 0; i < M; ++i) s += A[(size_t)i * C + j];
    out[j] = s;
}

// segment-mean accumulation: one block per edge, D = De + 2*Dx threads.
__global__ void scatter_accum(const float* __restrict__ e2, int De,
                              const float* __restrict__ x, int Dx,
                              const int* __restrict__ ei,
                              float* __restrict__ agg,
                              float* __restrict__ cnt, int D)
{
    const int m = blockIdx.x;
    const int s = ei[2 * m + 0];
    const int t = ei[2 * m + 1];
    const int j = threadIdx.x;
    float v;
    if (j < De)            v = e2[(size_t)m * De + j];
    else if (j < De + Dx)  v = x[(size_t)s * Dx + (j - De)];
    else                   v = x[(size_t)t * Dx + (j - De - Dx)];
    atomicAdd(&agg[(size_t)s * D + j], v);
    if (j == 0) atomicAdd(&cnt[s], 1.0f);
}

__global__ void div_mean(float* __restrict__ agg, const float* __restrict__ cnt,
                         size_t total, int D)
{
    size_t i = (size_t)blockIdx.x * blockDim.x + threadIdx.x;
    if (i >= total) return;
    agg[i] /= fmaxf(cnt[i / D], 1.0f);
}

// row softmax over 128 cols, one wave32 per row
__global__ __launch_bounds__(256)
void softmax128(const float* __restrict__ in, float* __restrict__ out, int rows)
{
    const int wave = threadIdx.x >> 5;
    const int lane = threadIdx.x & 31;
    const int row  = blockIdx.x * 8 + wave;
    if (row >= rows) return;
    const float* p = in + (size_t)row * 128;
    float v[4];
    float mx = -INFINITY;
#pragma unroll
    for (int j = 0; j < 4; ++j) { v[j] = p[lane + 32 * j]; mx = fmaxf(mx, v[j]); }
    for (int o = 16; o; o >>= 1) mx = fmaxf(mx, __shfl_xor(mx, o, 32));
    float sum = 0.0f;
#pragma unroll
    for (int j = 0; j < 4; ++j) { v[j] = __expf(v[j] - mx); sum += v[j]; }
    for (int o = 16; o; o >>= 1) sum += __shfl_xor(sum, o, 32);
    const float inv = 1.0f / sum;
    float* q = out + (size_t)row * 128;
#pragma unroll
    for (int j = 0; j < 4; ++j) q[lane + 32 * j] = v[j] * inv;
}

// ---------------------------------------------------------------------------
static inline Seg mkseg(const float* A, int lda, int K,
                        const int* ridx = nullptr, int rstride = 0)
{
    Seg s; s.A = A; s.ridx = ridx; s.rstride = rstride; s.lda = lda; s.K = K;
    return s;
}

static inline void launch_gemm(Seg a0, Seg a1, Seg a2, int nseg,
                               const float* W, const float* bias, float* C,
                               int M, int N, int flags, hipStream_t s)
{
    dim3 grid(M / 128, N / 64);
    gemm_bf16_wmma<<<grid, 256, 0, s>>>(a0, a1, a2, nseg, W, bias, C,
                                        M, N, flags);
}

extern "C" void kernel_launch(void* const* d_in, const int* in_sizes, int n_in,
                              void* d_out, int out_size, void* d_ws, size_t ws_size,
                              hipStream_t stream)
{
    const float* x  = (const float*)d_in[0];   // [N x 32]
    const float* e  = (const float*)d_in[1];   // [E x 16]
    const int*   ei = (const int*)  d_in[2];   // [E x 2] (src, tgt)
    const int Nn = in_sizes[0] / 32;
    const int E  = in_sizes[1] / 16;

    const float** P = (const float**)(d_in + 3);
    // 0..3  : emb_node  W0(32x256)  b0 W1(256x256) b1
    // 4..7  : emb_edge  W0(16x256)  b0 W1 b1
    // 8..11 : emb_glob  W0(1x256)   b0 W1 b1
    // 12..17: gnn0.edge W0(1024x256) b0 W1(256x256) b1 W2(256x128) b2
    // 18..23: gnn0.node W0(896x256)  b0 W1(256x256) b1 W2(256x128) b2
    // 24..29: gnn0.glob W0(512x256)  b0 W1(256x256) b1 W2(256x64)  b2
    // 30..31: gnn1.edge W0(448x128)  b0   (gnn1.node / gnn1.glob are dead:
    //                                      output softmax(e) only needs edge)

    float* wsf = (float*)d_ws;
    size_t off = 0;
    auto alloc = [&](size_t n) { float* p = wsf + off; off += n; return p; };
    float* eA   = alloc((size_t)E  * 256);
    float* eB   = alloc((size_t)E  * 256);
    float* nA   = alloc((size_t)Nn * 256);
    float* nB   = alloc((size_t)Nn * 256);
    float* agg  = alloc((size_t)Nn * 640);
    float* cnt  = alloc((size_t)Nn);
    float* u1   = alloc(256);
    float* uh   = alloc(256);
    float* u2   = alloc(64);
    float* gin  = alloc(512);
    float* gh1  = alloc(256);
    float* gh2  = alloc(256);
    float* beff = alloc(256);
    float* z1   = alloc(4);
    (void)ws_size;

    const Seg dummy = mkseg(nullptr, 0, 0);

    // ================= embeddings =================
    launch_gemm(mkseg(x, 32, 32), dummy, dummy, 1, P[0], P[1], nA,
                Nn, 256, GF_APPLY | GF_LEAKY, stream);
    launch_gemm(mkseg(nA, 256, 256), dummy, dummy, 1, P[2], P[3], nB,
                Nn, 256, GF_APPLY | GF_LEAKY, stream);
    launch_gemm(mkseg(e, 16, 16), dummy, dummy, 1, P[4], P[5], eA,
                E, 256, GF_APPLY | GF_LEAKY, stream);   // K=16 tail path
    launch_gemm(mkseg(eA, 256, 256), dummy, dummy, 1, P[6], P[7], eB,
                E, 256, GF_APPLY | GF_LEAKY, stream);
    // u: input is zeros(1,1)
    zero_f32<<<1, 64, 0, stream>>>(z1, 4);
    tiny_mlp<<<1, 256, 0, stream>>>(z1, 1,   P[8],  P[9],  uh, 256, 1);
    tiny_mlp<<<1, 256, 0, stream>>>(uh, 256, P[10], P[11], u1, 256, 1);

    // ================= gnn0: edge MLP =================
    // layer1 K=1024 = [e(256) | x[src](256) | x[tgt](256) | u(256, folded)]
    tiny_mlp<<<1, 256, 0, stream>>>(u1, 256, P[12] + (size_t)768 * 256, P[13],
                                    beff, 256, 0);
    launch_gemm(mkseg(eB, 256, 256),
                mkseg(nB, 256, 256, ei, 2),
                mkseg(nB, 256, 256, ei + 1, 2), 3,
                P[12], beff, eA, E, 256, GF_APPLY | GF_LEAKY, stream);
    launch_gemm(mkseg(eA, 256, 256), dummy, dummy, 1, P[14], P[15], eB,
                E, 256, GF_APPLY | GF_LEAKY, stream);
    launch_gemm(mkseg(eB, 256, 256), dummy, dummy, 1, P[16], P[17], eA,
                E, 128, GF_APPLY | GF_RELU, stream);    // e2 = eA (E x 128)

    // ================= gnn0: segment mean =================
    zero_f32<<<4096, 256, 0, stream>>>(agg, (size_t)Nn * 640);
    zero_f32<<<64,   256, 0, stream>>>(cnt, (size_t)Nn);
    scatter_accum<<<E, 640, 0, stream>>>(eA, 128, nB, 256, ei, agg, cnt, 640);
    {
        const size_t tot = (size_t)Nn * 640;
        div_mean<<<(unsigned)((tot + 255) / 256), 256, 0, stream>>>(agg, cnt,
                                                                    tot, 640);
    }

    // ================= gnn0: node MLP =================
    tiny_mlp<<<1, 256, 0, stream>>>(u1, 256, P[18] + (size_t)640 * 256, P[19],
                                    beff, 256, 0);
    launch_gemm(mkseg(agg, 640, 640), dummy, dummy, 1, P[18], beff, nA,
                Nn, 256, GF_APPLY | GF_LEAKY, stream);
    launch_gemm(mkseg(nA, 256, 256), dummy, dummy, 1, P[20], P[21], nB,
                Nn, 256, GF_APPLY | GF_LEAKY, stream);
    launch_gemm(mkseg(nB, 256, 256), dummy, dummy, 1, P[22], P[23], nA,
                Nn, 128, GF_APPLY | GF_RELU, stream);   // x2 = nA (N x 128)

    // ================= gnn0: global MLP =================
    colsum<<<1, 128, 0, stream>>>(nA, Nn, 128, gin);        // sum(x2)
    colsum<<<1, 128, 0, stream>>>(eA, E,  128, gin + 128);  // sum(e2)
    copy_f32<<<1, 256, 0, stream>>>(u1, gin + 256, 256);
    tiny_mlp<<<1, 256, 0, stream>>>(gin, 512, P[24], P[25], gh1, 256, 1);
    tiny_mlp<<<1, 256, 0, stream>>>(gh1, 256, P[26], P[27], gh2, 256, 1);
    tiny_mlp<<<1, 64,  0, stream>>>(gh2, 256, P[28], P[29], u2,  64,  2);

    // ================= gnn1: edge MLP (only live branch) =================
    tiny_mlp<<<1, 128, 0, stream>>>(u2, 64, P[30] + (size_t)384 * 128, P[31],
                                    beff, 128, 0);
    launch_gemm(mkseg(eA, 128, 128),
                mkseg(nA, 128, 128, ei, 2),
                mkseg(nA, 128, 128, ei + 1, 2), 3,
                P[30], beff, eB, E, 128, GF_APPLY | GF_RELU, stream);

    // ================= softmax -> d_out =================
    softmax128<<<E / 8, 256, 0, stream>>>(eB, (float*)d_out, E);
}
